// CrossAttention_1563368096520
// MI455X (gfx1250) — matrix operand order repaired and checked by reading, not compiled, not run on hardware
//
#include <hip/hip_runtime.h>
#include <hip/hip_bf16.h>

// ---------------------------------------------------------------------------
// CDNA5 / gfx1250 cross-attention: bf16 WMMA GEMMs + double-buffered flash
// attention with async global->LDS tile staging (ASYNCcnt path).
// Wave32 everywhere. Fragment layouts follow CDNA5 ISA:
//   A (16x32 bf16): lane L: row = L%16; elems 0..7  -> k = 8*(L/16)+0..7
//                                     elems 8..15 -> k = 16+8*(L/16)+0..7
//   B (32x16 bf16): lane L: col = L%16; elems 0..7 -> k = 8*(L/16)+0..7; 8..15 -> k = 16+8*(L/16)+0..7
//                   (B stored "N-major": Bt[n][k] contiguous in k)
//   C/D (16x16 f32): lane L: col = L%16; component i -> row = i + 8*(L/16)
// ---------------------------------------------------------------------------

typedef __attribute__((ext_vector_type(16))) __bf16 v16bf;
typedef __attribute__((ext_vector_type(8)))  __bf16 v8bf;
typedef __attribute__((ext_vector_type(8)))  float  v8f;

#define NB     2
#define SQ     4096
#define SKV    4096
#define QDIM   512
#define CDIM   768
#define NH     8
#define DH     64
#define INNER  512
#define KVT    64
#define SCALE_Q 0.125f   // 64^-0.5

union FragBF { v16bf v; v8bf h[2]; };

__device__ __forceinline__ v16bf load_a_frag(const __bf16* base, int ld, int row,
                                             int k0, int hi) {
  FragBF f;
  const __bf16* p = base + (size_t)row * ld + k0 + 8 * hi;
  f.h[0] = *(const v8bf*)(p);        // k = k0+8*hi .. +7
  f.h[1] = *(const v8bf*)(p + 16);   // k = k0+16+8*hi .. +7
  return f.v;
}

__device__ __forceinline__ v16bf load_b_frag(const __bf16* base, int ld, int n,
                                             int k0, int hi) {
  FragBF f;
  const __bf16* p = base + (size_t)n * ld + k0 + 8 * hi;
  f.h[0] = *(const v8bf*)(p);        // k = k0+8*hi .. +7
  f.h[1] = *(const v8bf*)(p + 16);   // k = k0+16+8*hi .. +7
  return f.v;
}

__device__ __forceinline__ v8f zero8() {
  v8f z = {0.f, 0.f, 0.f, 0.f, 0.f, 0.f, 0.f, 0.f};
  return z;
}

#define WMMA_BF16(a, b, c) \
  __builtin_amdgcn_wmma_f32_16x16x32_bf16(false, (a), false, (b), (short)0, (c), false, false)

// Async global -> LDS copy of 16 bytes per lane (GV addressing mode).
// ldsoff = low 32 bits of the generic LDS address (aperture mapping keeps the
// wave-relative LDS byte offset in addr[31:0]).
__device__ __forceinline__ void async_ld_b128(unsigned ldsoff, const void* gaddr) {
  asm volatile("global_load_async_to_lds_b128 %0, %1, off"
               :: "v"(ldsoff), "v"(gaddr)
               : "memory");
}

__device__ __forceinline__ void wait_async0() {
#if __has_builtin(__builtin_amdgcn_s_wait_asynccnt)
  __builtin_amdgcn_s_wait_asynccnt(0);
  asm volatile("" ::: "memory");
#else
  asm volatile("s_wait_asynccnt 0" ::: "memory");
#endif
}

// ---------------------------------------------------------------------------
// fp32 -> bf16 elementwise convert
// ---------------------------------------------------------------------------
__global__ void cvt_f32_bf16(const float* __restrict__ in, __bf16* __restrict__ out,
                             int n) {
  int i = blockIdx.x * 256 + threadIdx.x;
  if (i < n) out[i] = (__bf16)in[i];
}

// Transpose fp32 [R,C] -> bf16 [C,R]
__global__ void transpose_w(const float* __restrict__ in, __bf16* __restrict__ out,
                            int R, int C) {
  int i = blockIdx.x * 256 + threadIdx.x;
  if (i < R * C) {
    int r = i / C, c = i % C;
    out[(size_t)c * R + r] = (__bf16)in[i];
  }
}

// V [b*SKV, INNER] bf16 -> Vt [b, h, d, s] bf16 (per-head transposed)
__global__ void make_vt(const __bf16* __restrict__ Vb, __bf16* __restrict__ Vt) {
  int i = blockIdx.x * 256 + threadIdx.x;
  if (i < NB * SKV * INNER) {
    int c = i % INNER;
    int s = (i / INNER) % SKV;
    int b = i / (INNER * SKV);
    int h = c >> 6, d = c & 63;
    Vt[((size_t)((b * NH + h) * DH + d)) * SKV + s] = Vb[i];
  }
}

// ---------------------------------------------------------------------------
// bf16 WMMA GEMM: C[M,N] = scale * (A[M,K] x Bt[N,K]^T) + bias
// 128x128 block tile, 8 waves, each wave owns a 32x64 tile (2x4 WMMA tiles).
// Direct-from-global fragment loads (fully pipelined b128 + wmma, L2-resident
// operands).
// ---------------------------------------------------------------------------
__global__ __launch_bounds__(256)
void gemm_bf16_wmma(const __bf16* __restrict__ A, const __bf16* __restrict__ Bt,
                    void* __restrict__ Cout, const float* __restrict__ bias,
                    int M, int N, int K, float scale, int out_f32) {
  const int lane = threadIdx.x & 31;
  const int wave = threadIdx.x >> 5;
  const int r = lane & 15, hi = lane >> 4;
  const int nblk = N / 128;
  const int m0 = (blockIdx.x / nblk) * 128 + (wave >> 1) * 32;
  const int n0 = (blockIdx.x % nblk) * 128 + (wave & 1) * 64;

  v8f acc[2][4];
#pragma unroll
  for (int mt = 0; mt < 2; mt++)
#pragma unroll
    for (int t = 0; t < 4; t++) acc[mt][t] = zero8();

  for (int k0 = 0; k0 < K; k0 += 32) {
    v16bf a0 = load_a_frag(A, K, m0 + r, k0, hi);
    v16bf a1 = load_a_frag(A, K, m0 + 16 + r, k0, hi);
#pragma unroll
    for (int t = 0; t < 4; t++) {
      v16bf b = load_b_frag(Bt, K, n0 + t * 16 + r, k0, hi);
      acc[0][t] = WMMA_BF16(a0, b, acc[0][t]);
      acc[1][t] = WMMA_BF16(a1, b, acc[1][t]);
    }
  }

#pragma unroll
  for (int mt = 0; mt < 2; mt++)
#pragma unroll
    for (int t = 0; t < 4; t++)
#pragma unroll
      for (int i = 0; i < 8; i++) {
        int row = m0 + mt * 16 + 8 * hi + i;
        int col = n0 + t * 16 + r;
        float v = acc[mt][t][i] * scale;
        if (bias) v += bias[col];
        if (out_f32)
          ((float*)Cout)[(size_t)row * N + col] = v;
        else
          ((__bf16*)Cout)[(size_t)row * N + col] = (__bf16)v;
      }
}

// ---------------------------------------------------------------------------
// Flash attention: one block = 128 Q rows of one (b,h); 8 waves x 16 rows.
// Double-buffered KV tiles of 64 staged in LDS via async global->LDS loads;
// online softmax; O accumulated in fp32. Q pre-scaled by SCALE_Q.
// ---------------------------------------------------------------------------
__global__ __launch_bounds__(256)
void flash_attn(const __bf16* __restrict__ Qb, const __bf16* __restrict__ Kb,
                const __bf16* __restrict__ Vt, __bf16* __restrict__ Ob) {
  __shared__ __align__(16) __bf16 sK[2][KVT * DH];     // [kv][d]   2 x 8 KB
  __shared__ __align__(16) __bf16 sV[2][DH * KVT];     // [d][kv]   2 x 8 KB
  __shared__ __align__(16) __bf16 sP[8 * 16 * KVT];    // per-wave 16x64, 16 KB

  const int tid = threadIdx.x;
  const int lane = tid & 31, wave = tid >> 5;
  const int r = lane & 15, hi = lane >> 4;
  const int NQT = SQ / 128;
  const int qt = blockIdx.x % NQT;
  const int h = (blockIdx.x / NQT) % NH;
  const int b = blockIdx.x / (NQT * NH);
  const int q0 = qt * 128 + wave * 16;

  const __bf16* qbase = Qb + ((size_t)(b * SQ) + q0) * INNER + h * DH;
  const __bf16* kbase = Kb + (size_t)(b * SKV) * INNER + h * DH;
  const __bf16* vtbase = Vt + (size_t)((b * NH + h) * DH) * SKV;
  __bf16* pbuf = sP + wave * 16 * KVT;

  // Async stage of K tile [64 kv][64 d] and Vt tile [64 d][64 kv] into buf.
  // 512 x 16B chunks per tile, 256 threads -> 2 async b128 per tile each.
  auto stage = [&](int buf, int kv0) {
#pragma unroll
    for (int j = 0; j < 2; j++) {
      int c = tid + j * 256;
      int row = c >> 3, col8 = (c & 7) * 8;
      async_ld_b128((unsigned)(uintptr_t)(&sK[buf][row * DH + col8]),
                    kbase + (size_t)(kv0 + row) * INNER + col8);
      async_ld_b128((unsigned)(uintptr_t)(&sV[buf][row * KVT + col8]),
                    vtbase + (size_t)row * SKV + kv0 + col8);
    }
  };

  // Q fragments held in registers for the whole KV sweep (d-chunks 0..31, 32..63)
  v16bf qf[2];
  qf[0] = load_a_frag(qbase, INNER, r, 0, hi);
  qf[1] = load_a_frag(qbase, INNER, r, 32, hi);

  v8f o[4];
#pragma unroll
  for (int t = 0; t < 4; t++) o[t] = zero8();
  float m_i[8], l_i[8];
#pragma unroll
  for (int i = 0; i < 8; i++) { m_i[i] = -3.0e38f; l_i[i] = 0.f; }

  stage(0, 0);  // prefetch first tile

  for (int kv0 = 0; kv0 < SKV; kv0 += KVT) {
    const int cur = (kv0 >> 6) & 1;
    wait_async0();     // our async copies into sK/sV[cur] are complete
    __syncthreads();   // everyone's are complete; prev buffer fully consumed
    if (kv0 + KVT < SKV) stage(cur ^ 1, kv0 + KVT);  // overlap next tile fetch

    // scores S = Q . K^T : 4 N-tiles (kv), K-dim = d = 64 (2 chunks)
    v8f s[4];
#pragma unroll
    for (int t = 0; t < 4; t++) {
      v8f a = zero8();
      v16bf b0 = load_b_frag(sK[cur], DH, t * 16 + r, 0, hi);
      v16bf b1 = load_b_frag(sK[cur], DH, t * 16 + r, 32, hi);
      a = WMMA_BF16(qf[0], b0, a);
      a = WMMA_BF16(qf[1], b1, a);
      s[t] = a;
    }

    // Online softmax. Component i <-> row (i + 8*hi); columns live across the
    // 16-lane half, so xor-shuffles 1/2/4/8 reduce within each row.
    float mnew[8], alpha[8], rs[8];
#pragma unroll
    for (int i = 0; i < 8; i++) {
      float mx = s[0][i];
#pragma unroll
      for (int t = 1; t < 4; t++) mx = fmaxf(mx, s[t][i]);
#pragma unroll
      for (int d = 1; d < 16; d <<= 1) mx = fmaxf(mx, __shfl_xor(mx, d, 32));
      mnew[i] = fmaxf(m_i[i], mx);
      alpha[i] = __expf(m_i[i] - mnew[i]);
      rs[i] = 0.f;
    }
#pragma unroll
    for (int t = 0; t < 4; t++)
#pragma unroll
      for (int i = 0; i < 8; i++) {
        float p = __expf(s[t][i] - mnew[i]);
        rs[i] += p;
        pbuf[(8 * hi + i) * KVT + t * 16 + r] = (__bf16)p;
      }
#pragma unroll
    for (int i = 0; i < 8; i++) {
#pragma unroll
      for (int d = 1; d < 16; d <<= 1) rs[i] += __shfl_xor(rs[i], d, 32);
      l_i[i] = l_i[i] * alpha[i] + rs[i];
      m_i[i] = mnew[i];
    }
#pragma unroll
    for (int t = 0; t < 4; t++)
#pragma unroll
      for (int i = 0; i < 8; i++) o[t][i] *= alpha[i];

    __syncthreads();  // P visible in LDS before re-reading in A layout

    // O += P . V : A = P (16 x 64 kv), B = Vt tile, 4 d-tiles x 2 kv-chunks
#pragma unroll
    for (int kc = 0; kc < 2; kc++) {
      v16bf pa = load_a_frag(pbuf, KVT, r, kc * 32, hi);
#pragma unroll
      for (int t = 0; t < 4; t++) {
        v16bf vb = load_b_frag(sV[cur], KVT, t * 16 + r, kc * 32, hi);
        o[t] = WMMA_BF16(pa, vb, o[t]);
      }
    }
  }

  // Normalize and store O (bf16) back to [b*SQ, INNER] layout
  __bf16* obase = Ob + ((size_t)(b * SQ) + q0) * INNER + h * DH;
#pragma unroll
  for (int t = 0; t < 4; t++)
#pragma unroll
    for (int i = 0; i < 8; i++) {
      int row = 8 * hi + i;
      int col = t * 16 + r;
      obase[(size_t)row * INNER + col] = (__bf16)(o[t][i] / l_i[i]);
    }
}

// ---------------------------------------------------------------------------
// Host launcher
// ---------------------------------------------------------------------------
extern "C" void kernel_launch(void* const* d_in, const int* in_sizes, int n_in,
                              void* d_out, int out_size, void* d_ws, size_t ws_size,
                              hipStream_t stream) {
  const float* hidden = (const float*)d_in[0];  // [2,4096,512]
  const float* enc    = (const float*)d_in[1];  // [2,4096,768]
  const float* wq     = (const float*)d_in[2];  // [512,512]
  const float* wk     = (const float*)d_in[3];  // [768,512]
  const float* wv     = (const float*)d_in[4];  // [768,512]
  const float* wo     = (const float*)d_in[5];  // [512,512]
  const float* bo     = (const float*)d_in[6];  // [512]
  float* out = (float*)d_out;

  char* ws = (char*)d_ws;
  size_t off = 0;
  auto alloc = [&](size_t bytes) -> void* {
    void* p = ws + off;
    off += (bytes + 255) & ~(size_t)255;
    return p;
  };
  const size_t M = (size_t)NB * SQ;  // 8192
  __bf16* Hb  = (__bf16*)alloc(M * QDIM * 2);
  __bf16* Eb  = (__bf16*)alloc(M * CDIM * 2);
  __bf16* Wqt = (__bf16*)alloc((size_t)QDIM * INNER * 2);
  __bf16* Wkt = (__bf16*)alloc((size_t)CDIM * INNER * 2);
  __bf16* Wvt = (__bf16*)alloc((size_t)CDIM * INNER * 2);
  __bf16* Wot = (__bf16*)alloc((size_t)INNER * QDIM * 2);
  __bf16* Qb  = (__bf16*)alloc(M * INNER * 2);
  __bf16* Kb  = (__bf16*)alloc(M * INNER * 2);
  __bf16* Vb  = (__bf16*)alloc(M * INNER * 2);
  __bf16* Vt  = (__bf16*)alloc(M * INNER * 2);
  __bf16* Ob  = (__bf16*)alloc(M * INNER * 2);

  dim3 blk(256);
  int nH = (int)(M * QDIM), nE = (int)(M * CDIM);
  cvt_f32_bf16<<<(nH + 255) / 256, blk, 0, stream>>>(hidden, Hb, nH);
  cvt_f32_bf16<<<(nE + 255) / 256, blk, 0, stream>>>(enc, Eb, nE);
  transpose_w<<<(QDIM * INNER + 255) / 256, blk, 0, stream>>>(wq, Wqt, QDIM, INNER);
  transpose_w<<<(CDIM * INNER + 255) / 256, blk, 0, stream>>>(wk, Wkt, CDIM, INNER);
  transpose_w<<<(CDIM * INNER + 255) / 256, blk, 0, stream>>>(wv, Wvt, CDIM, INNER);
  transpose_w<<<(INNER * QDIM + 255) / 256, blk, 0, stream>>>(wo, Wot, INNER, QDIM);

  // Projections (scale folded into Q)
  int gQ = ((int)M / 128) * (INNER / 128);  // 64*4 = 256
  gemm_bf16_wmma<<<gQ, blk, 0, stream>>>(Hb, Wqt, Qb, nullptr, (int)M, INNER, QDIM,
                                         SCALE_Q, 0);
  gemm_bf16_wmma<<<gQ, blk, 0, stream>>>(Eb, Wkt, Kb, nullptr, (int)M, INNER, CDIM,
                                         1.0f, 0);
  gemm_bf16_wmma<<<gQ, blk, 0, stream>>>(Eb, Wvt, Vb, nullptr, (int)M, INNER, CDIM,
                                         1.0f, 0);

  int nV = (int)(M * INNER);
  make_vt<<<(nV + 255) / 256, blk, 0, stream>>>(Vb, Vt);

  flash_attn<<<NB * NH * (SQ / 128), blk, 0, stream>>>(Qb, Kb, Vt, Ob);

  // Output projection with bias, fp32 out
  gemm_bf16_wmma<<<gQ, blk, 0, stream>>>(Ob, Wot, (void*)out, bo, (int)M, QDIM, INNER,
                                         1.0f, 1);
}
